// Net_34857954575062
// MI455X (gfx1250) — matrix operand order repaired and checked by reading, not compiled, not run on hardware
//
#include <hip/hip_runtime.h>
#include <hip/hip_bf16.h>

// ---------------------------------------------------------------------------
// Progressive sparse local attention for MI455X (gfx1250, wave32, WMMA, TDM).
// Pass 1: bf16 WMMA GEMMs Et = Wf*Ft, Ee = Wg*Fte; results stored CHANNELS-
//         LAST bf16 ([b][hw][c]) in d_ws so pass-2 tiles are contiguous-row
//         2D copies for the Tensor Data Mover.
// Pass 2: TDM-staged (double-buffered, s_wait_tensorcnt) WMMA Q*K^T star
//         affinities, masked softmax, coalesced weighted gather of Ft.
// ---------------------------------------------------------------------------

typedef __attribute__((ext_vector_type(16))) __bf16 v16bf;
typedef __attribute__((ext_vector_type(8)))  float  v8f;
typedef __attribute__((ext_vector_type(4)))  unsigned int u32x4;
typedef __attribute__((ext_vector_type(4)))  int i32x4;
typedef __attribute__((ext_vector_type(8)))  int i32x8;

#define C_DIM 256
#define O_DIM 256
#define H_DIM 96
#define W_DIM 96
#define HW_DIM (H_DIM * W_DIM)
#define WS_GUARD 4096   // bytes before Et for left-edge TDM underflow

static __device__ __forceinline__ unsigned short f2bf(float x) {
  unsigned int u = __float_as_uint(x);
  unsigned int r = 0x7fffu + ((u >> 16) & 1u);   // round-to-nearest-even
  return (unsigned short)((u + r) >> 16);
}

// A fragment (16x32 bf16): lane L -> row M=L&15, K in {kb..kb+7}U{kb+16..kb+23}
static __device__ __forceinline__ v16bf load_fragA(const unsigned short* row, int lane) {
  const int kb = (lane >> 4) * 8;
  union { v16bf v; uint4 q[2]; } f;
  f.q[0] = *(const uint4*)(row + kb);
  f.q[1] = *(const uint4*)(row + kb + 16);
  return f.v;
}
// B fragment (32x16 bf16): lane L -> column N=L&15, K in {kb..kb+7}U{kb+16..kb+23}
static __device__ __forceinline__ v16bf load_fragB(const unsigned short* row, int lane) {
  const int kb = (lane >> 4) * 8;
  union { v16bf v; uint4 q[2]; } f;
  f.q[0] = *(const uint4*)(row + kb);
  f.q[1] = *(const uint4*)(row + kb + 16);
  return f.v;
}

#if __has_builtin(__builtin_amdgcn_tensor_load_to_lds)
#define USE_TDM 1
// 2D TDM load: tile_d0 contiguous elements (2B each) x tile_d1 rows with a
// row stride of stride0 elements; destination LDS is tile-contiguous.
static __device__ __forceinline__ void tdm_load_2d(unsigned lds_off, const void* gptr,
                                                   unsigned tile_d0, unsigned tile_d1,
                                                   unsigned stride0) {
  unsigned long long ga = (unsigned long long)(uintptr_t)gptr;
  u32x4 g0 = { 1u,                                  // count=1 (valid user D#)
               lds_off,                             // lds_addr (bytes)
               (unsigned)ga,                        // global_addr[31:0]
               (unsigned)(ga >> 32) | (2u << 30) }; // global_addr[56:32], type=2
  i32x8 g1 = { (int)0x00010000u,                    // data_size=2B; no multicast
               0,                                   // no atomic barrier; dim0 lo16=0
               (int)0x00004000u,                    // tensor_dim0=2^30, dim1 lo16=0
               (int)(0x00004000u | (tile_d0 << 16)),// tensor_dim1=2^30, tile_dim0
               (int)tile_d1,                        // tile_dim1 (tile_dim2=0)
               (int)stride0,                        // tensor_dim0_stride lo32
               0, 0 };
  i32x4 gz = {0, 0, 0, 0};
#if defined(__clang_major__) && __clang_major__ >= 23
  i32x8 gz8 = {0, 0, 0, 0, 0, 0, 0, 0};
  __builtin_amdgcn_tensor_load_to_lds(g0, g1, gz, gz, gz8, 0);
#else
  __builtin_amdgcn_tensor_load_to_lds(g0, g1, gz, gz, 0);
#endif
}
#else
#define USE_TDM 0
#endif

// ---------------------------------------------------------------------------
// Pass 1: embedding GEMM.  grid = (HW/64, O/64, B*2), block = 128 (4 waves).
// Output channels-last bf16: E[b][hw][o].
// ---------------------------------------------------------------------------
__global__ __launch_bounds__(128) void embed_gemm_kernel(
    const float* __restrict__ Ft, const float* __restrict__ Fe,
    const float* __restrict__ Wf, const float* __restrict__ bfv,
    const float* __restrict__ Wg, const float* __restrict__ bgv,
    unsigned short* __restrict__ Et, unsigned short* __restrict__ Ee)
{
  __shared__ alignas(16) unsigned short As[64][32];   // [o][c] chunk (bf16)
  __shared__ alignas(16) unsigned short Bs[64][32];   // [px][c] chunk (bf16)

  const int tid  = threadIdx.x;
  const int lane = tid & 31;
  const int wv   = tid >> 5;
  const int hw0  = blockIdx.x * 64;
  const int o0   = blockIdx.y * 64;
  const int b    = blockIdx.z >> 1;
  const int e    = blockIdx.z & 1;

  const float* F  = (e ? Fe : Ft) + (size_t)b * C_DIM * HW_DIM;
  const float* Wm = e ? Wg : Wf;
  const float* bv = e ? bgv : bfv;
  unsigned short* E = (e ? Ee : Et) + (size_t)b * HW_DIM * C_DIM;  // channels-last

  v8f acc[4] = {{}, {}, {}, {}};

  for (int k0 = 0; k0 < C_DIM; k0 += 32) {
    __syncthreads();
#pragma unroll
    for (int i = 0; i < 4; ++i) {            // weights 64x32 -> As[o][c]
      int idx = tid + 128 * i;
      int row = idx >> 3, c4 = (idx & 7) * 4;
      float4 v = *(const float4*)&Wm[(size_t)(o0 + row) * C_DIM + k0 + c4];
      As[row][c4 + 0] = f2bf(v.x);
      As[row][c4 + 1] = f2bf(v.y);
      As[row][c4 + 2] = f2bf(v.z);
      As[row][c4 + 3] = f2bf(v.w);
    }
#pragma unroll
    for (int i = 0; i < 4; ++i) {            // features 32c x 64px -> Bs[px][c]
      int idx = tid + 128 * i;
      int row = idx >> 4, p4 = (idx & 15) * 4;
      float4 v = *(const float4*)&F[(size_t)(k0 + row) * HW_DIM + hw0 + p4];
      Bs[p4 + 0][row] = f2bf(v.x);
      Bs[p4 + 1][row] = f2bf(v.y);
      Bs[p4 + 2][row] = f2bf(v.z);
      Bs[p4 + 3][row] = f2bf(v.w);
    }
    __syncthreads();

    if (k0 + 32 < C_DIM)                     // global_prefetch_b8 next chunk
      __builtin_prefetch(&F[(size_t)(k0 + 32 + lane) * HW_DIM + hw0 + wv * 16], 0, 0);

    v16bf af = load_fragA(&As[wv * 16 + (lane & 15)][0], lane);
#pragma unroll
    for (int n = 0; n < 4; ++n) {
      v16bf bfg = load_fragB(&Bs[n * 16 + (lane & 15)][0], lane);
      acc[n] = __builtin_amdgcn_wmma_f32_16x16x32_bf16(
          false, af, false, bfg, (short)0, acc[n], false, false);
    }
  }

  // Epilogue: lane holds 8 consecutive O values of one pixel record ->
  // one 16-byte global_store_b128 per tile (channels-last).
  const int n0 = lane & 15;
  const int ob = o0 + wv * 16 + (lane >> 4) * 8;
  float4 b0 = *(const float4*)&bv[ob];
  float4 b1 = *(const float4*)&bv[ob + 4];
#pragma unroll
  for (int n = 0; n < 4; ++n) {
    int px = hw0 + n * 16 + n0;
    unsigned p0 = f2bf(acc[n][0] + b0.x) | ((unsigned)f2bf(acc[n][1] + b0.y) << 16);
    unsigned p1 = f2bf(acc[n][2] + b0.z) | ((unsigned)f2bf(acc[n][3] + b0.w) << 16);
    unsigned p2 = f2bf(acc[n][4] + b1.x) | ((unsigned)f2bf(acc[n][5] + b1.y) << 16);
    unsigned p3 = f2bf(acc[n][6] + b1.z) | ((unsigned)f2bf(acc[n][7] + b1.w) << 16);
    uint4 q = {p0, p1, p2, p3};
    *(uint4*)&E[(size_t)px * C_DIM + ob] = q;
  }
}

// ---------------------------------------------------------------------------
// Star-neighborhood tables (offset order matches reference _offsets()).
// ---------------------------------------------------------------------------
__device__ const signed char KTAB[9][9] = {
  {25,-1,-1,-1,26,-1,-1,-1,27},
  {-1,17,-1,-1,18,-1,-1,19,-1},
  {-1,-1, 9,-1,10,-1,11,-1,-1},
  {-1,-1,-1, 1, 2, 3,-1,-1,-1},
  {28,20,12, 4, 0, 5,13,21,29},
  {-1,-1,-1, 6, 7, 8,-1,-1,-1},
  {-1,-1,14,-1,15,-1,16,-1,-1},
  {-1,22,-1,-1,23,-1,-1,24,-1},
  {30,-1,-1,-1,31,-1,-1,-1,32},
};
__device__ const signed char AK[33] = {0,-1,-1,-1,0,0,1,1,1,-2,-2,-2,0,0,2,2,2,
                                       -3,-3,-3,0,0,3,3,3,-4,-4,-4,0,0,4,4,4};
__device__ const signed char BK[33] = {0,-1,0,1,-1,1,-1,0,1,-2,0,2,-2,2,-2,0,2,
                                       -3,0,3,-3,3,-3,0,3,-4,0,4,-4,4,-4,0,4};

// ---------------------------------------------------------------------------
// Pass 2: fused attention.  grid = (W/32, H, B), block = 256 (8 waves).
// TDM stages Q (32px x 32c) and 9 key halo rows (40px x 32c) per chunk,
// double-buffered; 18 WMMA jobs over 8 waves compute the star affinities.
// ---------------------------------------------------------------------------
__global__ __launch_bounds__(256) void attn_kernel(
    const float* __restrict__ Ft, const unsigned short* __restrict__ Et,
    const unsigned short* __restrict__ Ee, float* __restrict__ out)
{
  __shared__ alignas(16) unsigned short Qs[2][32][32];      // [buf][p][c]
  __shared__ alignas(16) unsigned short Ks[2][9][40][32];   // [buf][a][q][c]
  __shared__ float Sc[32][33];                              // scores -> weights

  const int tid  = threadIdx.x;
  const int lane = tid & 31;
  const int wv   = tid >> 5;
  const int w0   = blockIdx.x * 32;
  const int h    = blockIdx.y;
  const int b    = blockIdx.z;

  const unsigned short* Eeb = Ee + (size_t)b * HW_DIM * C_DIM;  // channels-last
  const unsigned short* Etb = Et + (size_t)b * HW_DIM * C_DIM;

  v8f zero = {};
  v8f acc[3][2];
#pragma unroll
  for (int jj = 0; jj < 3; ++jj) { acc[jj][0] = zero; acc[jj][1] = zero; }

#if USE_TDM
  // Issue all 10 tile descriptors of one 32-channel chunk (wave 0 only).
  auto issue_chunk = [&](int cc, int buf) {
    const int c0 = cc * 32;
    tdm_load_2d((unsigned)(uintptr_t)&Qs[buf][0][0],
                Eeb + (size_t)(h * W_DIM + w0) * C_DIM + c0, 32, 32, C_DIM);
#pragma unroll
    for (int s = 0; s < 9; ++s) {
      int hh = min(max(h + s - 4, 0), H_DIM - 1);   // out-of-range w is masked
      tdm_load_2d((unsigned)(uintptr_t)&Ks[buf][s][0][0],
                  Etb + (ptrdiff_t)(hh * W_DIM + w0 - 4) * C_DIM + c0, 32, 40, C_DIM);
    }
  };
  if (wv == 0) issue_chunk(0, 0);
#endif

  for (int cc = 0; cc < 8; ++cc) {
    const int buf = cc & 1;
#if USE_TDM
    if (wv == 0) {
      if (cc < 7) {                       // pipeline: issue next, wait current
        issue_chunk(cc + 1, buf ^ 1);
        __builtin_amdgcn_s_wait_tensorcnt(10);
      } else {
        __builtin_amdgcn_s_wait_tensorcnt(0);
      }
    }
#else
    {  // manual staging fallback (channels-last: contiguous c-runs)
      const int c0 = cc * 32;
      int p = tid >> 3, l8 = tid & 7;
      *(unsigned long long*)&Qs[buf][p][l8 * 4] =
          *(const unsigned long long*)(Eeb + (size_t)(h * W_DIM + w0 + p) * C_DIM + c0 + l8 * 4);
#pragma unroll
      for (int i = 0; i < 12; ++i) {
        int idx = tid + 256 * i;
        if (idx < 2880) {
          int row = idx >> 3, j8 = (idx & 7) * 4;
          int s = row / 40, q = row % 40;
          int hh = min(max(h + s - 4, 0), H_DIM - 1);
          int ww = min(max(w0 - 4 + q, 0), W_DIM - 1);
          *(unsigned long long*)&Ks[buf][s][q][j8] =
              *(const unsigned long long*)(Etb + (size_t)(hh * W_DIM + ww) * C_DIM + c0 + j8);
        }
      }
    }
#endif
    __syncthreads();   // chunk cc staged for all waves

#pragma unroll
    for (int jj = 0; jj < 3; ++jj) {
      int j = wv + jj * 8;
      if (j < 18) {
        int ai = j >> 1, m = j & 1;
        v16bf af = load_fragA(&Qs[buf][m * 16 + (lane & 15)][0], lane);
#pragma unroll
        for (int t = 0; t < 2; ++t) {
          int qoff = m * 16 + t * 8;      // N-tile window offsets {0,8}/{16,24}
          v16bf bfg = load_fragB(&Ks[buf][ai][qoff + (lane & 15)][0], lane);
          acc[jj][t] = __builtin_amdgcn_wmma_f32_16x16x32_bf16(
              false, af, false, bfg, (short)0, acc[jj][t], false, false);
        }
      }
    }
    __syncthreads();   // compute on buf done before it is refilled
  }

  // Extract star entries: D elem (M,N) -> p = m*16 + r + hi*8, q = qoff + n
  {
    const int n  = lane & 15;
    const int mh = (lane >> 4) * 8;
#pragma unroll
    for (int jj = 0; jj < 3; ++jj) {
      int j = wv + jj * 8;
      if (j < 18) {
        int ai = j >> 1, m = j & 1;
#pragma unroll
        for (int t = 0; t < 2; ++t) {
          int qoff = m * 16 + t * 8;
#pragma unroll
          for (int r = 0; r < 8; ++r) {
            int p  = m * 16 + r + mh;
            int b0 = qoff + n - 4 - p;
            if (b0 >= -4 && b0 <= 4) {
              int k = KTAB[ai][b0 + 4];
              if (k >= 0) Sc[p][k] = acc[jj][t][r];
            }
          }
        }
      }
    }
  }
  __syncthreads();

  // Masked softmax over the 33 offsets (one thread per query).
  if (tid < 32) {
    const int p = tid, wq = w0 + p;
    float sv[33], mx = -3.0e38f;
#pragma unroll
    for (int k = 0; k < 33; ++k) {
      int hh = h + AK[k], ww = wq + BK[k];
      bool ok = (unsigned)hh < (unsigned)H_DIM && (unsigned)ww < (unsigned)W_DIM;
      float s = ok ? Sc[p][k] : -1.0e30f;
      sv[k] = s;
      mx = fmaxf(mx, s);
    }
    float sum = 0.f;
#pragma unroll
    for (int k = 0; k < 33; ++k) { sv[k] = __expf(sv[k] - mx); sum += sv[k]; }
    float inv = 1.f / sum;
#pragma unroll
    for (int k = 0; k < 33; ++k) Sc[p][k] = sv[k] * inv;
  }
  __syncthreads();

  // Weighted gather of Ft: lanes <-> w (coalesced), 8 channel groups.
  {
    const int p  = tid & 31;
    const int cg = tid >> 5;
    const int wq = w0 + p;
    float wk[33];
    int   off[33];
#pragma unroll
    for (int k = 0; k < 33; ++k) {
      wk[k] = Sc[p][k];
      int hh = min(max(h + AK[k], 0), H_DIM - 1);
      int ww = min(max(wq + BK[k], 0), W_DIM - 1);
      off[k] = hh * W_DIM + ww;   // weight is exactly 0 when clamped-invalid
    }
    const float* Fb = Ft + (size_t)b * C_DIM * HW_DIM;
    float* Ob = out + (size_t)b * C_DIM * HW_DIM + h * W_DIM + wq;
    for (int ci = 0; ci < 32; ++ci) {
      int c = cg + ci * 8;
      const float* Fc = Fb + (size_t)c * HW_DIM;
      float a = 0.f;
#pragma unroll
      for (int k = 0; k < 33; ++k) a = fmaf(wk[k], Fc[off[k]], a);
      Ob[(size_t)c * HW_DIM] = a;
    }
  }
}

// ---------------------------------------------------------------------------
extern "C" void kernel_launch(void* const* d_in, const int* in_sizes, int n_in,
                              void* d_out, int out_size, void* d_ws, size_t ws_size,
                              hipStream_t stream) {
  const float* Ft  = (const float*)d_in[0];
  const float* Fte = (const float*)d_in[1];
  const float* Wf  = (const float*)d_in[2];
  const float* bfv = (const float*)d_in[3];
  const float* Wg  = (const float*)d_in[4];
  const float* bgv = (const float*)d_in[5];
  float* out = (float*)d_out;

  const size_t elems = (size_t)4 * HW_DIM * C_DIM;      // per-embedding bf16 elems
  unsigned short* Et = (unsigned short*)((char*)d_ws + WS_GUARD);  // guard band
  unsigned short* Ee = Et + elems;                      // 37.7 MB total + guard

  dim3 g1(HW_DIM / 64, O_DIM / 64, 4 * 2), b1(128);
  embed_gemm_kernel<<<g1, b1, 0, stream>>>(Ft, Fte, Wf, bfv, Wg, bgv, Et, Ee);

  dim3 g2(W_DIM / 32, H_DIM, 4), b2(256);
  attn_kernel<<<g2, b2, 0, stream>>>(Ft, Et, Ee, out);
}
